// GRU_55241869361382
// MI455X (gfx1250) — matrix operand order repaired and checked by reading, not compiled, not run on hardware
//
#include <hip/hip_runtime.h>
#include <hip/hip_bf16.h>

// GRU on MI455X (gfx1250): f16 WMMA (v_wmma_f32_16x16x32_f16) everywhere,
// f32 accumulation. Phase 1 = big parallel input GEMM; phase 2 = 512 serial
// steps with K-split-over-waves + LDS reduction to cut per-step latency.

typedef __attribute__((ext_vector_type(16))) _Float16 v16h;
typedef __attribute__((ext_vector_type(8)))  _Float16 v8h;
typedef __attribute__((ext_vector_type(8)))  float    v8f;

#define T_STEPS 512
#define B_SZ    64
#define I_SZ    1024
#define H_SZ    1024
#define O_SZ    1024
#define G3H     (3 * H_SZ)

union HalfVec { v16h v; v8h h[2]; };

// A fragment (16x32, M x K), A row-major with row pitch ldk (halves).
// ISA layout: lanes 0-15 hold M=lane, elems 0..7 = K 0..7, elems 8..15 = K 16..23;
// lanes 16-31 hold M=lane-16, elems 0..7 = K 8..15, elems 8..15 = K 24..31.
__device__ __forceinline__ v16h load_a_frag(const _Float16* __restrict__ A, int ldk) {
    const int lane = threadIdx.x & 31;
    const int row  = lane & 15;
    const int k0   = (lane >> 4) << 3;   // 0 or 8
    const _Float16* p = A + (size_t)row * ldk + k0;
    HalfVec r;
    r.h[0] = *reinterpret_cast<const v8h*>(p);
    r.h[1] = *reinterpret_cast<const v8h*>(p + 16);
    return r.v;
}

// B fragment (32x16, K x N) where B = W^T, W row-major [N_total][ldk].
// Column n of B is row n of W -> two runs of 8 contiguous halves per lane.
// Layout: lane&15 = column, lanes 0-15 hold K 0..7,16..23, lanes 16-31 hold K 8..15,24..31.
__device__ __forceinline__ v16h load_b_frag(const _Float16* __restrict__ W,
                                            int n0, int kk, int ldk) {
    const int lane = threadIdx.x & 31;
    const _Float16* p = W + (size_t)(n0 + (lane & 15)) * ldk + kk + ((lane >> 4) << 3);
    HalfVec r;
    r.h[0] = *reinterpret_cast<const v8h*>(p);
    r.h[1] = *reinterpret_cast<const v8h*>(p + 16);
    return r.v;
}

__device__ __forceinline__ v8f wmma16(v16h a, v16h b, v8f c) {
    return __builtin_amdgcn_wmma_f32_16x16x32_f16(false, a, false, b, (short)0, c,
                                                  false, false);
}

__device__ __forceinline__ float sigmoidf_fast(float x) {
    return 1.0f / (1.0f + __expf(-x));
}

// ---------------------------------------------------------------- utilities
__global__ void cvt_f32_to_f16(const float* __restrict__ src,
                               _Float16* __restrict__ dst, int n) {
    int i = blockIdx.x * blockDim.x + threadIdx.x;
    const int stride = gridDim.x * blockDim.x;
    for (; i < n; i += stride) dst[i] = (_Float16)src[i];
}

__global__ void zero_h_kernel(float* __restrict__ hf, _Float16* __restrict__ h16, int n) {
    int i = blockIdx.x * blockDim.x + threadIdx.x;
    if (i < n) { hf[i] = 0.0f; h16[i] = (_Float16)0.0f; }
}

// ------------------------------------------------- phase 1: gi = X @ Wih^T + b_ih
// X: [T*B, I] f16 row-major, Wih: [3H, I] f16 row-major, gi: [T*B, 3H] f16.
// One wave computes a 16x64 output tile (4 accumulators, A fragment reused 4x).
__global__ void __launch_bounds__(256) gi_gemm_kernel(
    const _Float16* __restrict__ X, const _Float16* __restrict__ Wih,
    const float* __restrict__ bih, _Float16* __restrict__ gi) {
    const int wave = (blockIdx.x << 3) + (threadIdx.x >> 5);
    const int NW_N = G3H / 64;               // 48 n-blocks
    const int mt = wave / NW_N;              // 0..2047
    const int n0 = (wave % NW_N) * 64;
    const _Float16* Arow = X + (size_t)mt * 16 * I_SZ;

    v8f acc[4] = {};
    for (int kk = 0; kk < I_SZ; kk += 32) {
        v16h a = load_a_frag(Arow + kk, I_SZ);
#pragma unroll
        for (int j = 0; j < 4; ++j) {
            v16h b = load_b_frag(Wih, n0 + j * 16, kk, I_SZ);
            acc[j] = wmma16(a, b, acc[j]);
        }
    }
    const int lane = threadIdx.x & 31;
    const int nloc = lane & 15;
    const int mofs = (lane >> 4) << 3;
#pragma unroll
    for (int j = 0; j < 4; ++j) {
        const int n = n0 + j * 16 + nloc;
        const float bv = bih[n];
#pragma unroll
        for (int i = 0; i < 8; ++i) {
            const int m = mt * 16 + mofs + i;
            gi[(size_t)m * G3H + n] = (_Float16)(acc[j][i] + bv);
        }
    }
}

// ------------------------------- phase 2a: fused recurrent step (gh + gates + h_new)
// Block = 4 waves cooperating on one 16(batch) x 64(h) tile; K=1024 split 4-way
// across waves (8 WMMA k-steps each), partials reduced through LDS.
// Grid = 4 (m-tiles) * 16 (h-blocks) = 64 blocks.
__global__ void __launch_bounds__(128) gru_step_kernel(
    const _Float16* __restrict__ Hprev16, const float* __restrict__ HprevF,
    const _Float16* __restrict__ Whh, const float* __restrict__ bhh,
    const _Float16* __restrict__ gi_t,          // [B, 3H] slice at t
    float* __restrict__ HnewF, _Float16* __restrict__ Hnew16) {
    __shared__ float part[4][3][4][32][8];      // 48 KB: [wave][gate][j][lane][i]

    const int w    = threadIdx.x >> 5;
    const int lane = threadIdx.x & 31;
    const int mt   = blockIdx.x >> 4;           // 0..3
    const int n0   = (blockIdx.x & 15) * 64;    // h block base
    const _Float16* Arow = Hprev16 + (size_t)mt * 16 * H_SZ;

    v8f acc[3][4] = {};
    const int kbeg = w * (H_SZ / 4);
    for (int kk = kbeg; kk < kbeg + H_SZ / 4; kk += 32) {
        v16h a = load_a_frag(Arow + kk, H_SZ);
#pragma unroll
        for (int g = 0; g < 3; ++g) {
#pragma unroll
            for (int j = 0; j < 4; ++j) {
                v16h b = load_b_frag(Whh, g * H_SZ + n0 + j * 16, kk, H_SZ);
                acc[g][j] = wmma16(a, b, acc[g][j]);
            }
        }
    }
#pragma unroll
    for (int g = 0; g < 3; ++g)
#pragma unroll
        for (int j = 0; j < 4; ++j)
#pragma unroll
            for (int i = 0; i < 8; ++i) part[w][g][j][lane][i] = acc[g][j][i];
    __syncthreads();

    // wave w finalizes sub-tile j == w
    const int nloc = lane & 15;
    const int mofs = (lane >> 4) << 3;
    const int h    = n0 + w * 16 + nloc;
    const float br = bhh[h], bz = bhh[H_SZ + h], bn = bhh[2 * H_SZ + h];
#pragma unroll
    for (int i = 0; i < 8; ++i) {
        const int b = mt * 16 + mofs + i;
        float gr = br, gz = bz, gn = bn;
#pragma unroll
        for (int w2 = 0; w2 < 4; ++w2) {
            gr += part[w2][0][w][lane][i];
            gz += part[w2][1][w][lane][i];
            gn += part[w2][2][w][lane][i];
        }
        const float xr = (float)gi_t[(size_t)b * G3H + h];
        const float xz = (float)gi_t[(size_t)b * G3H + H_SZ + h];
        const float xn = (float)gi_t[(size_t)b * G3H + 2 * H_SZ + h];
        const float r  = sigmoidf_fast(xr + gr);
        const float z  = sigmoidf_fast(xz + gz);
        const float nn = tanhf(xn + r * gn);
        const float hp = HprevF[(size_t)b * H_SZ + h];
        const float hn = (1.0f - z) * nn + z * hp;
        HnewF[(size_t)b * H_SZ + h]  = hn;
        Hnew16[(size_t)b * H_SZ + h] = (_Float16)hn;
    }
}

// --------------------------------- phase 2b: y = h_new @ Wout^T + b_out (per step)
// Same K-split-4 structure. Grid = 64 blocks of 128 threads.
__global__ void __launch_bounds__(128) out_gemm_kernel(
    const _Float16* __restrict__ Hn16, const _Float16* __restrict__ Wout,
    const float* __restrict__ bout, float* __restrict__ Y) {
    __shared__ float part[4][4][32][8];         // 16 KB

    const int w    = threadIdx.x >> 5;
    const int lane = threadIdx.x & 31;
    const int mt   = blockIdx.x >> 4;
    const int n0   = (blockIdx.x & 15) * 64;
    const _Float16* Arow = Hn16 + (size_t)mt * 16 * H_SZ;

    v8f acc[4] = {};
    const int kbeg = w * (H_SZ / 4);
    for (int kk = kbeg; kk < kbeg + H_SZ / 4; kk += 32) {
        v16h a = load_a_frag(Arow + kk, H_SZ);
#pragma unroll
        for (int j = 0; j < 4; ++j) {
            v16h b = load_b_frag(Wout, n0 + j * 16, kk, H_SZ);
            acc[j] = wmma16(a, b, acc[j]);
        }
    }
#pragma unroll
    for (int j = 0; j < 4; ++j)
#pragma unroll
        for (int i = 0; i < 8; ++i) part[w][j][lane][i] = acc[j][i];
    __syncthreads();

    const int nloc = lane & 15;
    const int mofs = (lane >> 4) << 3;
    const int n    = n0 + w * 16 + nloc;
    const float bv = bout[n];
#pragma unroll
    for (int i = 0; i < 8; ++i) {
        const int m = mt * 16 + mofs + i;
        float s = bv;
#pragma unroll
        for (int w2 = 0; w2 < 4; ++w2) s += part[w2][w][lane][i];
        Y[(size_t)m * O_SZ + n] = s;
    }
}

extern "C" void kernel_launch(void* const* d_in, const int* in_sizes, int n_in,
                              void* d_out, int out_size, void* d_ws, size_t ws_size,
                              hipStream_t stream) {
    const float* inputs = (const float*)d_in[0];  // [T,B,I]
    const float* w_ih   = (const float*)d_in[1];  // [3H,I]
    const float* w_hh   = (const float*)d_in[2];  // [3H,H]
    const float* b_ih   = (const float*)d_in[3];  // [3H]
    const float* b_hh   = (const float*)d_in[4];  // [3H]
    const float* w_out  = (const float*)d_in[5];  // [O,H]
    const float* b_out  = (const float*)d_in[6];  // [O]
    float* out = (float*)d_out;                   // [T*B*O] outputs ++ [B*H] h_last

    // ---- workspace carve (all f16 buffers 256B-aligned by construction)
    char* p = (char*)d_ws;
    _Float16* X16    = (_Float16*)p; p += (size_t)T_STEPS * B_SZ * I_SZ * 2;
    _Float16* Wih16  = (_Float16*)p; p += (size_t)G3H * I_SZ * 2;
    _Float16* Whh16  = (_Float16*)p; p += (size_t)G3H * H_SZ * 2;
    _Float16* Wout16 = (_Float16*)p; p += (size_t)O_SZ * H_SZ * 2;
    _Float16* gi     = (_Float16*)p; p += (size_t)T_STEPS * B_SZ * G3H * 2;
    float*    hF0    = (float*)p;    p += (size_t)B_SZ * H_SZ * 4;
    float*    hF1    = (float*)p;    p += (size_t)B_SZ * H_SZ * 4;
    _Float16* h160   = (_Float16*)p; p += (size_t)B_SZ * H_SZ * 2;
    _Float16* h161   = (_Float16*)p; p += (size_t)B_SZ * H_SZ * 2;
    float*    hF[2]    = { hF0, hF1 };
    _Float16* h16[2]   = { h160, h161 };

    // ---- f32 -> f16 conversions
    {
        const int nX = T_STEPS * B_SZ * I_SZ;
        cvt_f32_to_f16<<<8192, 256, 0, stream>>>(inputs, X16, nX);
        cvt_f32_to_f16<<<4096, 256, 0, stream>>>(w_ih, Wih16, G3H * I_SZ);
        cvt_f32_to_f16<<<4096, 256, 0, stream>>>(w_hh, Whh16, G3H * H_SZ);
        cvt_f32_to_f16<<<2048, 256, 0, stream>>>(w_out, Wout16, O_SZ * H_SZ);
    }

    // ---- phase 1: gi = X @ Wih^T + b_ih  (2048 m-tiles x 48 n-blocks, 8 waves/block)
    gi_gemm_kernel<<<(2048 * 48) / 8, 256, 0, stream>>>(X16, Wih16, b_ih, gi);

    // ---- h0 = 0
    zero_h_kernel<<<(B_SZ * H_SZ) / 256, 256, 0, stream>>>(hF[0], h16[0], B_SZ * H_SZ);

    // ---- phase 2: 512 recurrent steps (ping-pong h buffers)
    for (int t = 0; t < T_STEPS; ++t) {
        const int cur = t & 1, nxt = cur ^ 1;
        gru_step_kernel<<<64, 128, 0, stream>>>(
            h16[cur], hF[cur], Whh16, b_hh,
            gi + (size_t)t * B_SZ * G3H, hF[nxt], h16[nxt]);
        out_gemm_kernel<<<64, 128, 0, stream>>>(
            h16[nxt], Wout16, b_out, out + (size_t)t * B_SZ * O_SZ);
    }

    // ---- h_last (after step T-1, state lives in buffer index T&1 == 0)
    hipMemcpyAsync(out + (size_t)T_STEPS * B_SZ * O_SZ, hF[0],
                   (size_t)B_SZ * H_SZ * sizeof(float),
                   hipMemcpyDeviceToDevice, stream);
}
